// WindowAttention_85942295593071
// MI455X (gfx1250) — matrix operand order repaired and checked by reading, not compiled, not run on hardware
//
#include <hip/hip_runtime.h>
#include <hip/hip_bf16.h>

typedef _Float16 half_t;
typedef __attribute__((ext_vector_type(16))) _Float16 v16h;
typedef __attribute__((ext_vector_type(8)))  _Float16 v8h;
typedef __attribute__((ext_vector_type(8)))  float    v8f;

// ---------------- problem constants ----------------
#define NTOK 49     // tokens per window
#define NPAD 64     // padded tokens (4 tiles of 16)
#define CDIM 384
#define NHEAD 12
#define HDIM 32
#define NQKV 1152   // 3*C
#define XS   392    // xh row stride (halves): 784B, 16B multiple, 4-dword bank skew
#define PS   72     // P row stride (halves): 144B, 16B multiple

// LDS layout (bytes)
#define OFF_X 0
#define SZ_X  (NPAD * XS * 2)                 // 50176
#define OFF_Q (OFF_X + SZ_X)
#define SZ_Q  (NHEAD * NPAD * HDIM * 2)       // 49152
#define OFF_K (OFF_Q + SZ_Q)
#define OFF_V (OFF_K + SZ_Q)
#define OFF_P (OFF_V + SZ_Q)
#define SZ_P  (NHEAD * 16 * PS * 2)           // 27648
#define SMEM_BYTES (OFF_P + SZ_P)             // 225280 (<320KB/WGP)

// ---------------- WMMA helpers ----------------
// A fragment (16x32 f16, MxK): lane half 0 -> K {0..7,16..23}, half 1 -> K {8..15,24..31}.
// p must already point at row*stride + khalf_base; elements 0..7 = p[0..7], 8..15 = p[16..23].
static __device__ __forceinline__ v16h load_a(const half_t* p) {
    v8h lo = *(const v8h*)(p);
    v8h hi = *(const v8h*)(p + 16);
    return __builtin_shufflevector(lo, hi, 0,1,2,3,4,5,6,7,8,9,10,11,12,13,14,15);
}
// B fragment (32x16 f16, KxN, stored [N][K] contiguous-K): lane half 0 -> K {0..7,16..23}, half 1 -> K {8..15,24..31}.
static __device__ __forceinline__ v16h load_b(const half_t* p) {
    v8h lo = *(const v8h*)(p);
    v8h hi = *(const v8h*)(p + 16);
    return __builtin_shufflevector(lo, hi, 0,1,2,3,4,5,6,7,8,9,10,11,12,13,14,15);
}
static __device__ __forceinline__ v8f wmma32(v16h a, v16h b, v8f c) {
    return __builtin_amdgcn_wmma_f32_16x16x32_f16(false, a, false, b, (short)0, c, false, false);
}

// ---------------- weight fp32 -> f16 conversion ----------------
__global__ void WinAttn_cvt_weights(const float* __restrict__ qkv_w,
                                    const float* __restrict__ proj_w,
                                    half_t* __restrict__ qwh,
                                    half_t* __restrict__ pwh) {
    const int nq = NQKV * CDIM;
    const int np = CDIM * CDIM;
    for (int i = blockIdx.x * blockDim.x + threadIdx.x; i < nq + np;
         i += gridDim.x * blockDim.x) {
        if (i < nq) qwh[i] = (half_t)qkv_w[i];
        else        pwh[i - nq] = (half_t)proj_w[i - nq];
    }
}

// ---------------- fused window attention ----------------
__global__ __launch_bounds__(384, 1)
void WinAttn_fused(const float* __restrict__ x,
                   const float* __restrict__ mask,
                   const float* __restrict__ qkv_b,
                   const float* __restrict__ proj_b,
                   const float* __restrict__ rpb,
                   const half_t* __restrict__ qwh,
                   const half_t* __restrict__ pwh,
                   float* __restrict__ out) {
    extern __shared__ char smem[];
    half_t* axh = (half_t*)(smem + OFF_X);  // phase1-2: x f16 [64][392]; phase3-4: attn-out
    half_t* qh  = (half_t*)(smem + OFF_Q);  // [12][64][32]
    half_t* kh  = (half_t*)(smem + OFF_K);  // [12][64][32]
    half_t* vT  = (half_t*)(smem + OFF_V);  // [12][32][64]  (v transposed)
    half_t* Pws = (half_t*)(smem + OFF_P);  // [12][16][72]  per-wave P tile

    const int b     = blockIdx.x;
    const int tid   = threadIdx.x;
    const int wave  = tid >> 5;
    const int lane  = tid & 31;
    const int lrow  = lane & 15;
    const int lhalf = lane >> 4;

    // ---- phase 1: stage x[b] into LDS as f16, zero-pad rows 49..63 ----
    {
        const float* xb = x + (size_t)b * NTOK * CDIM;
        for (int i = tid; i < NPAD * CDIM; i += 384) {
            int n = i / CDIM, c = i - n * CDIM;
            axh[n * XS + c] = (n < NTOK) ? (half_t)xb[n * CDIM + c] : (half_t)0.0f;
        }
    }
    __syncthreads();

    // ---- phase 2: QKV GEMM  [64 x 1152] = xh[64 x 384] @ qkv_w^T ----
    for (int t = wave; t < 4 * (NQKV / 16); t += 12) {
        const int mt = t & 3, nt = t >> 2;
        const float bias = qkv_b[nt * 16 + lrow];
        v8f acc = {bias, bias, bias, bias, bias, bias, bias, bias};
        const half_t* arow = axh + (mt * 16 + lrow) * XS + lhalf * 8;
        const half_t* brow = qwh + (size_t)(nt * 16 + lrow) * CDIM + lhalf * 8;
#pragma unroll
        for (int kk = 0; kk < CDIM / 32; ++kk)
            acc = wmma32(load_a(arow + kk * 32), load_b(brow + kk * 32), acc);

        const int sec = nt / 24;                 // 0=q 1=k 2=v (uniform per wave)
        const int cc  = nt * 16 + lrow - sec * CDIM;
        const int hh  = cc >> 5, d = cc & 31;
#pragma unroll
        for (int r = 0; r < 8; ++r) {
            const int row = mt * 16 + r + 8 * lhalf;
            const half_t v = (half_t)acc[r];
            if (sec == 0)      qh[(hh * NPAD + row) * HDIM + d] = v;
            else if (sec == 1) kh[(hh * NPAD + row) * HDIM + d] = v;
            else               vT[(hh * HDIM + d) * NPAD + row] = v;
        }
    }
    __syncthreads();

    // ---- phase 3: attention, one head per wave ----
    {
        const int h = wave;
        const float* maskb = mask + (size_t)(b & 63) * NTOK * NTOK;
        const half_t* qbase = qh + h * NPAD * HDIM;
        const half_t* kbase = kh + h * NPAD * HDIM;
        const half_t* vbase = vT + h * HDIM * NPAD;
        half_t* Pw = Pws + h * 16 * PS;
        const float scale = 0.17677669529663687f;   // 1/sqrt(32)

#pragma unroll 1
        for (int mt = 0; mt < 4; ++mt) {
            // S row-tile: 4 column tiles
            float sv[4][8];
#pragma unroll
            for (int nt = 0; nt < 4; ++nt) {
                v8f acc = {};
                v16h a = load_a(qbase + (mt * 16 + lrow) * HDIM + lhalf * 8);
                v16h bb = load_b(kbase + (nt * 16 + lrow) * HDIM + lhalf * 8);
                acc = wmma32(a, bb, acc);
                const int m = nt * 16 + lrow;
#pragma unroll
                for (int r = 0; r < 8; ++r) {
                    const int n = mt * 16 + r + 8 * lhalf;
                    float s = acc[r] * scale;
                    if (n < NTOK && m < NTOK) {
                        int rn0 = n / 7, rn1 = n - rn0 * 7;
                        int rm0 = m / 7, rm1 = m - rm0 * 7;
                        int ridx = (rn0 - rm0 + 6) * 13 + (rn1 - rm1 + 6);
                        s += rpb[ridx * NHEAD + h] + maskb[n * NTOK + m];
                    } else {
                        s = -1e30f;  // mask padded columns (and garbage rows)
                    }
                    sv[nt][r] = s;
                }
            }
            // row softmax (rows live in 16-lane halves -> xor-butterfly 1,2,4,8)
            float rs[8];
#pragma unroll
            for (int r = 0; r < 8; ++r) {
                float mx = fmaxf(fmaxf(sv[0][r], sv[1][r]), fmaxf(sv[2][r], sv[3][r]));
                mx = fmaxf(mx, __shfl_xor(mx, 1));
                mx = fmaxf(mx, __shfl_xor(mx, 2));
                mx = fmaxf(mx, __shfl_xor(mx, 4));
                mx = fmaxf(mx, __shfl_xor(mx, 8));
                float sum = 0.0f;
                const int prow = r + 8 * lhalf;
#pragma unroll
                for (int nt = 0; nt < 4; ++nt) {
                    float p = __expf(sv[nt][r] - mx);
                    sum += p;
                    Pw[prow * PS + nt * 16 + lrow] = (half_t)p;
                }
                sum += __shfl_xor(sum, 1);
                sum += __shfl_xor(sum, 2);
                sum += __shfl_xor(sum, 4);
                sum += __shfl_xor(sum, 8);
                rs[r] = sum;
            }
            __syncthreads();  // uniform (4 iters in every wave); orders P stores vs loads

            // out row-tile = P(16x64) @ V(64x32); divide by row-sum afterwards
#pragma unroll
            for (int ntd = 0; ntd < 2; ++ntd) {
                v8f acc = {};
#pragma unroll
                for (int kk = 0; kk < 2; ++kk) {
                    v16h a = load_a(Pw + lrow * PS + kk * 32 + lhalf * 8);
                    v16h bb = load_b(vbase + (ntd * 16 + lrow) * NPAD + kk * 32 + lhalf * 8);
                    acc = wmma32(a, bb, acc);
                }
                const int col = h * HDIM + ntd * 16 + lrow;
#pragma unroll
                for (int r = 0; r < 8; ++r) {
                    const int row = mt * 16 + r + 8 * lhalf;
                    axh[row * XS + col] = (half_t)(acc[r] / rs[r]);
                }
            }
        }
    }
    __syncthreads();

    // ---- phase 4: output projection  out[49x384] = attn_out @ proj_w^T + b ----
    for (int t = wave; t < 4 * (CDIM / 16); t += 12) {
        const int mt = t & 3, nt = t >> 2;
        const float bias = proj_b[nt * 16 + lrow];
        v8f acc = {bias, bias, bias, bias, bias, bias, bias, bias};
        const half_t* arow = axh + (mt * 16 + lrow) * XS + lhalf * 8;
        const half_t* brow = pwh + (size_t)(nt * 16 + lrow) * CDIM + lhalf * 8;
#pragma unroll
        for (int kk = 0; kk < CDIM / 32; ++kk)
            acc = wmma32(load_a(arow + kk * 32), load_b(brow + kk * 32), acc);
#pragma unroll
        for (int r = 0; r < 8; ++r) {
            const int row = mt * 16 + r + 8 * lhalf;
            if (row < NTOK)
                out[(size_t)b * NTOK * CDIM + row * CDIM + nt * 16 + lrow] = acc[r];
        }
    }
}

extern "C" void kernel_launch(void* const* d_in, const int* in_sizes, int n_in,
                              void* d_out, int out_size, void* d_ws, size_t ws_size,
                              hipStream_t stream) {
    const float* x      = (const float*)d_in[0];
    const float* mask   = (const float*)d_in[1];
    const float* qkv_w  = (const float*)d_in[2];
    const float* qkv_b  = (const float*)d_in[3];
    const float* proj_w = (const float*)d_in[4];
    const float* proj_b = (const float*)d_in[5];
    const float* rpb    = (const float*)d_in[6];
    float* out = (float*)d_out;

    half_t* qwh = (half_t*)d_ws;                 // [1152][384] f16
    half_t* pwh = qwh + (size_t)NQKV * CDIM;     // [384][384]  f16

    WinAttn_cvt_weights<<<dim3(2304), dim3(256), 0, stream>>>(qkv_w, proj_w, qwh, pwh);

    (void)hipFuncSetAttribute((const void*)WinAttn_fused,
                              hipFuncAttributeMaxDynamicSharedMemorySize, SMEM_BYTES);
    WinAttn_fused<<<dim3(2048), dim3(384), SMEM_BYTES, stream>>>(
        x, mask, qkv_b, proj_b, rpb, qwh, pwh, out);
}
